// MultiHeadAttention_25151328485613
// MI455X (gfx1250) — compile-verified, matrix operand order checked by reading, not run on hardware
//
#include <hip/hip_runtime.h>

#define BATCH 4
#define SEQ   2048
#define DIMC  1024
#define HEADS 16
#define HD    64
#define ROWS  (BATCH*SEQ)   // 8192

typedef _Float16 half8  __attribute__((ext_vector_type(8)));
typedef _Float16 half16 __attribute__((ext_vector_type(16)));
typedef float    float8 __attribute__((ext_vector_type(8)));
typedef float    float4v __attribute__((ext_vector_type(4)));
typedef unsigned uint32x4 __attribute__((ext_vector_type(4)));
typedef int      int32x4  __attribute__((ext_vector_type(4)));
typedef int      int32x8  __attribute__((ext_vector_type(8)));

#if __has_builtin(__builtin_amdgcn_tensor_load_to_lds)
#define USE_TDM 1
#endif

union H16U { half16 f; half8 h[2]; };

// Load one 16-half WMMA fragment strip for this lane.
// p points at (row-of-operand)*stride + k0 ; kb = (lane>>4)*8.
// Halves kb..kb+7 and kb+16..kb+23 per the gfx1250 16-bit A/B VGPR layout.
__device__ __forceinline__ half16 frag_ld(const _Float16* p, int kb) {
  H16U u;
  u.h[0] = *(const half8*)(p + kb);
  u.h[1] = *(const half8*)(p + kb + 16);
  return u.f;
}

__device__ __forceinline__ float8 wmma_f16(half16 a, half16 b, float8 c) {
  return __builtin_amdgcn_wmma_f32_16x16x32_f16(false, a, false, b, (short)0, c,
                                                false, false);
}

// VALU cross-lane max-reduce across the 16 lanes of each half-wave using
// v_permlane16_b32 butterflies (XOR 1,2,4,8 nibble selectors). No LDS traffic.
template <unsigned SL, unsigned SH>
__device__ __forceinline__ float plane16(float x) {
  unsigned u = __builtin_bit_cast(unsigned, x);
  unsigned r = __builtin_amdgcn_permlane16(u, u, SL, SH, false, false);
  return __builtin_bit_cast(float, r);
}
__device__ __forceinline__ float row16_max(float x) {
  x = fmaxf(x, plane16<0x67452301u, 0xEFCDAB89u>(x));  // xor 1
  x = fmaxf(x, plane16<0x54761032u, 0xDCFE98BAu>(x));  // xor 2
  x = fmaxf(x, plane16<0x32107654u, 0xBA98FEDCu>(x));  // xor 4
  x = fmaxf(x, plane16<0xFEDCBA98u, 0x76543210u>(x));  // xor 8
  return x;
}

#ifdef USE_TDM
// Issue a TDM 2-D tile load: tileH rows of tileW f16 elements, row stride
// rowStrideElems, into contiguous LDS at ldsAddr (bytes). D# per CDNA5 ISA
// ch.8: group0 = {flags, lds_addr, global_addr, type=2}; group1 = dims/strides.
__device__ __forceinline__ void tdm_load_2d(const _Float16* gsrc, unsigned ldsAddr,
                                            unsigned tileW, unsigned tileH,
                                            unsigned long long rowStrideElems) {
  unsigned long long ga = (unsigned long long)(size_t)gsrc;
  uint32x4 g0;
  g0[0] = 1u;                                   // count=1 (valid user D#)
  g0[1] = ldsAddr;                              // LDS byte address
  g0[2] = (unsigned)ga;                         // global_addr[31:0]
  g0[3] = (unsigned)((ga >> 32) & 0x01ffffffu)  // global_addr[56:32]
          | 0x80000000u;                        // type=2 ("image")
  int32x8 g1;
  g1[0] = (int)(1u << 16);                      // data_size=1 -> 2 bytes/elem
  g1[1] = (int)((tileW & 0xffffu) << 16);       // tensor_dim0[15:0]
  g1[2] = (int)(((tileW >> 16) & 0xffffu) |     // tensor_dim0[31:16]
                ((tileH & 0xffffu) << 16));     // tensor_dim1[15:0]
  g1[3] = (int)(((tileH >> 16) & 0xffffu) |     // tensor_dim1[31:16]
                ((tileW & 0xffffu) << 16));     // tile_dim0
  g1[4] = (int)(tileH & 0xffffu);               // tile_dim1 (tile_dim2=0 -> 2D)
  g1[5] = (int)(unsigned)(rowStrideElems & 0xffffffffu);  // dim0_stride[31:0]
  g1[6] = (int)(unsigned)((rowStrideElems >> 32) & 0xffffu); // dim0_stride[47:32]
  g1[7] = 0;
  int32x4 z4 = {0, 0, 0, 0};
  int32x8 z8 = {0, 0, 0, 0, 0, 0, 0, 0};
  __builtin_amdgcn_tensor_load_to_lds(g0, g1, z4, z4, z8, 0);
}
#endif

// ---------------------------------------------------------------- converts
// 8 elements per thread: two b128 loads + one b128 store.
__global__ __launch_bounds__(256) void cvt_f32_f16_v8(const float* __restrict__ in,
                                                      _Float16* __restrict__ out,
                                                      int n) {
  int i = (blockIdx.x * 256 + threadIdx.x) * 8;
  const int stride = gridDim.x * 256 * 8;
  for (; i < n; i += stride) {
    float4v a = *(const float4v*)(in + i);
    float4v c = *(const float4v*)(in + i + 4);
    half8 o;
    o[0] = (_Float16)a[0]; o[1] = (_Float16)a[1];
    o[2] = (_Float16)a[2]; o[3] = (_Float16)a[3];
    o[4] = (_Float16)c[0]; o[5] = (_Float16)c[1];
    o[6] = (_Float16)c[2]; o[7] = (_Float16)c[3];
    *(half8*)(out + i) = o;
  }
}

// ---------------------------------------------------------------- NT GEMM
// Y[M,N] = A[M,K] * Bw[N,K]^T + bias, M=8192, N=K=1024.
// Register double-buffered: iteration k+1's fragment loads issue before
// iteration k's WMMAs so global latency overlaps the matrix pipe.
// MODE 0: f16 head layout [B,H,SEQ,HD];  MODE 1: f16 V-transposed [B,H,HD,SEQ];
// MODE 3: f32 row-major [M,N] (final output).
template <int MODE>
__global__ __launch_bounds__(128) void gemm_nt(const _Float16* __restrict__ A,
                                               const _Float16* __restrict__ Bw,
                                               const float* __restrict__ bias,
                                               void* __restrict__ outp) {
  const int lane = threadIdx.x & 31;
  const int wv   = threadIdx.x >> 5;
  const int bx   = blockIdx.x;
  const int rowBlk = bx >> 4;            // 128 blocks of 64 rows
  const int colBlk = bx & 15;            // 16 blocks of 64 cols
  const int row0 = rowBlk * 64 + wv * 16;
  const int col0 = colBlk * 64;
  const int ml   = lane & 15;
  const int kb8  = (lane >> 4) * 8;
  const int rhalf = (lane >> 4) * 8;

  const _Float16* arow  = A  + (size_t)(row0 + ml) * DIMC;
  const _Float16* brow0 = Bw + (size_t)(col0 + ml) * DIMC;
  const _Float16* brow1 = brow0 + (size_t)16 * DIMC;
  const _Float16* brow2 = brow0 + (size_t)32 * DIMC;
  const _Float16* brow3 = brow0 + (size_t)48 * DIMC;

  float8 acc0 = {}, acc1 = {}, acc2 = {}, acc3 = {};

  half16 aC  = frag_ld(arow,  kb8);
  half16 b0C = frag_ld(brow0, kb8);
  half16 b1C = frag_ld(brow1, kb8);
  half16 b2C = frag_ld(brow2, kb8);
  half16 b3C = frag_ld(brow3, kb8);

  for (int k0 = 0; k0 < DIMC; k0 += 32) {
    const int kn = (k0 + 32) & (DIMC - 1);   // wraps to 0 on last iter (in-bounds)
    half16 aN  = frag_ld(arow + kn,  kb8);
    half16 b0N = frag_ld(brow0 + kn, kb8);
    half16 b1N = frag_ld(brow1 + kn, kb8);
    half16 b2N = frag_ld(brow2 + kn, kb8);
    half16 b3N = frag_ld(brow3 + kn, kb8);
    acc0 = wmma_f16(aC, b0C, acc0);
    acc1 = wmma_f16(aC, b1C, acc1);
    acc2 = wmma_f16(aC, b2C, acc2);
    acc3 = wmma_f16(aC, b3C, acc3);
    aC = aN; b0C = b0N; b1C = b1N; b2C = b2N; b3C = b3N;
  }

  float8 accs[4] = {acc0, acc1, acc2, acc3};
#pragma unroll
  for (int t = 0; t < 4; ++t) {
    const int n = col0 + 16 * t + ml;
    const float bvv = bias[n];
#pragma unroll
    for (int r = 0; r < 8; ++r) {
      const int mrow = row0 + r + rhalf;
      const float val = accs[t][r] + bvv;
      if (MODE == 0) {
        const int b = mrow >> 11, i = mrow & (SEQ - 1);
        const int h = n >> 6, d = n & (HD - 1);
        ((_Float16*)outp)[(((size_t)(b * HEADS + h)) * SEQ + i) * HD + d] =
            (_Float16)val;
      } else if (MODE == 1) {
        const int b = mrow >> 11, i = mrow & (SEQ - 1);
        const int h = n >> 6, d = n & (HD - 1);
        ((_Float16*)outp)[(((size_t)(b * HEADS + h)) * HD + d) * SEQ + i] =
            (_Float16)val;
      } else {
        ((float*)outp)[(size_t)mrow * DIMC + n] = val;
      }
    }
  }
}

// ---------------------------------------------------------------- attention
// Flash-style, block-cooperative, double-buffered: the 64x64 K tile and V^T
// tile for step j+1 are fetched by the Tensor Data Mover (one
// tensor_load_to_lds each, issued by waves 0/1, tracked on TENSORcnt) while
// all 4 waves run step j's WMMAs/softmax. One barrier per step.
// Row-max via v_permlane16 butterflies; row-sum via WMMA against ones.
__global__ __launch_bounds__(128) void attn_kernel(const _Float16* __restrict__ q,
                                                   const _Float16* __restrict__ k,
                                                   const _Float16* __restrict__ vt,
                                                   _Float16* __restrict__ ao) {
  __shared__ __align__(16) _Float16 kT[2][64 * 64];   // [buf][j_local][64]
  __shared__ __align__(16) _Float16 vT[2][64 * 64];   // [buf][d][j_local]
  __shared__ __align__(16) _Float16 pTile[4 * 16 * 64];
  const int tid  = threadIdx.x;
  const int lane = tid & 31;
  const int wv   = tid >> 5;
  const int bx = blockIdx.x;
  const int bh    = bx >> 5;           // b*HEADS + h
  const int itile = bx & 31;           // 32 row-tiles of 64
  const int i0 = itile * 64 + wv * 16;
  const int b = bh >> 4, h = bh & (HEADS - 1);
  const int ml    = lane & 15;
  const int kb8   = (lane >> 4) * 8;
  const int rhalf = (lane >> 4) * 8;

  const _Float16* qbase = q  + ((size_t)bh * SEQ + i0 + ml) * HD;
  const _Float16* kbase = k  + (size_t)bh * SEQ * HD;   // contiguous [SEQ][64]
  const _Float16* vbase = vt + (size_t)bh * HD * SEQ;   // [64][SEQ]
  _Float16* myP = pTile + wv * (16 * 64);

  // stage one j-step's K/V tiles into LDS buffer `buf`
  auto stage = [&](int buf, int j0) {
#ifdef USE_TDM
    if (wv == 0)
      tdm_load_2d(kbase + (size_t)j0 * HD,
                  (unsigned)(size_t)&kT[buf][0], 64, 64, 64);
    else if (wv == 1)
      tdm_load_2d(vbase + j0,
                  (unsigned)(size_t)&vT[buf][0], 64, 64, SEQ);
#else
#pragma unroll
    for (int c = 0; c < 4; ++c) {
      const int e = tid + 128 * c;               // 0..511 b128 chunks
      *(half8*)(&kT[buf][0] + e * 8) =
          *(const half8*)(kbase + (size_t)j0 * HD + e * 8);
      const int d = e >> 3, c8 = e & 7;          // V^T row d, 8-half chunk c8
      *(half8*)(&vT[buf][0] + d * 64 + c8 * 8) =
          *(const half8*)(vbase + (size_t)d * SEQ + j0 + c8 * 8);
    }
#endif
  };

  // Q fragments, softmax scale 1/sqrt(64)=0.125 folded in (exact pow2 in f16)
  half16 aq0 = frag_ld(qbase, kb8);
  half16 aq1 = frag_ld(qbase + 32, kb8);
  const _Float16 qs = (_Float16)0.125f;
#pragma unroll
  for (int e = 0; e < 16; ++e) { aq0[e] *= qs; aq1[e] *= qs; }

  half16 ones;
#pragma unroll
  for (int e = 0; e < 16; ++e) ones[e] = (_Float16)1.0f;

  float8 o0 = {}, o1 = {}, o2 = {}, o3 = {};
  float mrun[8], lrun[8];
#pragma unroll
  for (int r = 0; r < 8; ++r) { mrun[r] = -1e30f; lrun[r] = 0.0f; }

  // preload tile 0
  stage(0, 0);
#ifdef USE_TDM
  __builtin_amdgcn_s_wait_tensorcnt(0);
#endif
  __syncthreads();

  const int NJT = SEQ / 64;
  for (int jt = 0; jt < NJT; ++jt) {
    const int cur = jt & 1;
    const _Float16* kbuf = &kT[cur][0];
    const _Float16* vbuf = &vT[cur][0];

    // ---- fragments from LDS
    half16 kf[4][2], vf[4][2];
#pragma unroll
    for (int t = 0; t < 4; ++t) {
      const _Float16* krow = kbuf + (16 * t + ml) * 64;
      kf[t][0] = frag_ld(krow, kb8);
      kf[t][1] = frag_ld(krow + 32, kb8);
      const _Float16* vrow = vbuf + (16 * t + ml) * 64;
      vf[t][0] = frag_ld(vrow, kb8);
      vf[t][1] = frag_ld(vrow + 32, kb8);
    }
    // ---- kick off the NEXT tile's DMA into the other buffer
    if (jt + 1 < NJT) stage(cur ^ 1, (jt + 1) * 64);

    // ---- S = Q*K^T (8 WMMA)
    float8 s[4];
#pragma unroll
    for (int t = 0; t < 4; ++t) {
      float8 z = {};
      z = wmma_f16(aq0, kf[t][0], z);
      z = wmma_f16(aq1, kf[t][1], z);
      s[t] = z;
    }
    // ---- row-wise max (VALU permlane butterflies), rescale factors
    float mnew[8], alpha[8];
#pragma unroll
    for (int r = 0; r < 8; ++r) {
      float cm = fmaxf(fmaxf(s[0][r], s[1][r]), fmaxf(s[2][r], s[3][r]));
      cm = row16_max(cm);
      mnew[r]  = fmaxf(mrun[r], cm);
      alpha[r] = __expf(mrun[r] - mnew[r]);
      mrun[r]  = mnew[r];
    }
    // ---- P = exp(S - m) -> f16 -> per-wave LDS (row-major 16x64)
#pragma unroll
    for (int t = 0; t < 4; ++t)
#pragma unroll
      for (int r = 0; r < 8; ++r)
        myP[(r + rhalf) * 64 + 16 * t + ml] =
            (_Float16)__expf(s[t][r] - mnew[r]);
    // ---- rescale O while the LDS stores land
#pragma unroll
    for (int r = 0; r < 8; ++r) {
      o0[r] *= alpha[r]; o1[r] *= alpha[r];
      o2[r] *= alpha[r]; o3[r] *= alpha[r];
    }
    half16 pa0 = frag_ld(myP + ml * 64, kb8);
    half16 pa1 = frag_ld(myP + ml * 64 + 32, kb8);
    // ---- row-sum = P @ ones (WMMA, fp32 accumulate) + O += P*V
    float8 rsv = {};
    rsv = wmma_f16(pa0, ones, rsv);
    rsv = wmma_f16(pa1, ones, rsv);
    o0 = wmma_f16(pa0, vf[0][0], o0);  o0 = wmma_f16(pa1, vf[0][1], o0);
    o1 = wmma_f16(pa0, vf[1][0], o1);  o1 = wmma_f16(pa1, vf[1][1], o1);
    o2 = wmma_f16(pa0, vf[2][0], o2);  o2 = wmma_f16(pa1, vf[2][1], o2);
    o3 = wmma_f16(pa0, vf[3][0], o3);  o3 = wmma_f16(pa1, vf[3][1], o3);
#pragma unroll
    for (int r = 0; r < 8; ++r) lrun[r] = lrun[r] * alpha[r] + rsv[r];

    if (jt + 1 < NJT) {
#ifdef USE_TDM
      __builtin_amdgcn_s_wait_tensorcnt(0);
#endif
      __syncthreads();   // next tile resident; prev reads finished everywhere
    }
  }
  // ---- normalize and store into [B*SEQ, DIMC] f16 for the output projection
  float inv[8];
#pragma unroll
  for (int r = 0; r < 8; ++r) inv[r] = 1.0f / lrun[r];
  float8 oArr[4] = {o0, o1, o2, o3};
#pragma unroll
  for (int t = 0; t < 4; ++t)
#pragma unroll
    for (int r = 0; r < 8; ++r) {
      const int tok = b * SEQ + i0 + r + rhalf;
      const int ch  = h * HD + 16 * t + ml;
      ao[(size_t)tok * DIMC + ch] = (_Float16)(oArr[t][r] * inv[r]);
    }
}

// ---------------------------------------------------------------- launcher
extern "C" void kernel_launch(void* const* d_in, const int* in_sizes, int n_in,
                              void* d_out, int out_size, void* d_ws, size_t ws_size,
                              hipStream_t stream) {
  (void)in_sizes; (void)n_in; (void)out_size; (void)ws_size;
  const float* x  = (const float*)d_in[0];
  const float* Wq = (const float*)d_in[1];
  const float* bq = (const float*)d_in[2];
  const float* Wk = (const float*)d_in[3];
  const float* bk = (const float*)d_in[4];
  const float* Wv = (const float*)d_in[5];
  const float* bv = (const float*)d_in[6];
  const float* Wo = (const float*)d_in[7];
  const float* bo = (const float*)d_in[8];
  float* out = (float*)d_out;

  const size_t NX = (size_t)ROWS * DIMC;   // 8,388,608 halves
  const size_t NW = (size_t)DIMC * DIMC;   // 1,048,576 halves
  _Float16* ws = (_Float16*)d_ws;
  size_t off = 0;
  _Float16* xh  = ws + off; off += NX;
  _Float16* wqh = ws + off; off += NW;
  _Float16* wkh = ws + off; off += NW;
  _Float16* wvh = ws + off; off += NW;
  _Float16* woh = ws + off; off += NW;
  _Float16* qh  = ws + off; off += NX;
  _Float16* kh  = ws + off; off += NX;
  _Float16* vth = ws + off; off += NX;
  _Float16* aoh = ws + off; off += NX;

  cvt_f32_f16_v8<<<4096, 256, 0, stream>>>(x,  xh,  (int)NX);
  cvt_f32_f16_v8<<<512, 256, 0, stream>>>(Wq, wqh, (int)NW);
  cvt_f32_f16_v8<<<512, 256, 0, stream>>>(Wk, wkh, (int)NW);
  cvt_f32_f16_v8<<<512, 256, 0, stream>>>(Wv, wvh, (int)NW);
  cvt_f32_f16_v8<<<512, 256, 0, stream>>>(Wo, woh, (int)NW);

  gemm_nt<0><<<2048, 128, 0, stream>>>(xh, wqh, bq, (void*)qh);
  gemm_nt<0><<<2048, 128, 0, stream>>>(xh, wkh, bk, (void*)kh);
  gemm_nt<1><<<2048, 128, 0, stream>>>(xh, wvh, bv, (void*)vth);

  attn_kernel<<<2048, 128, 0, stream>>>(qh, kh, vth, aoh);

  gemm_nt<3><<<2048, 128, 0, stream>>>(aoh, woh, bo, (void*)out);
}
